// VanillaRNN_23063974380055
// MI455X (gfx1250) — matrix operand order repaired and checked by reading, not compiled, not run on hardware
//
#include <hip/hip_runtime.h>
#include <math.h>

// ---------------------------------------------------------------------------
// VanillaRNN on MI455X (gfx1250, wave32, WMMA)
//   B=64, T=512, N=1024
//   out[b,t,:] = tanh(h[b,t-1] @ W_self + X[b,t,:] @ W_input^T), W_self symmetric
// ---------------------------------------------------------------------------

#define N_DIM   1024
#define T_DIM   512
#define B_DIM   64

typedef __attribute__((ext_vector_type(16))) __bf16 v16bf;
typedef __attribute__((ext_vector_type(8)))  float  v8f;
typedef int v4i __attribute__((vector_size(16)));

struct __align__(16) u4x { unsigned x, y, z, w; };
struct __align__(32) U32x8 { u4x lo, hi; };

// ---- CDNA5 async global->LDS path (guarded; falls back to plain loads) ----
#if defined(__has_builtin)
#  if __has_builtin(__builtin_amdgcn_global_load_async_to_lds_b128) && \
      __has_builtin(__builtin_amdgcn_s_wait_asynccnt)
#    define HAVE_ASYNC_LDS 1
#  endif
#  if __has_builtin(__builtin_amdgcn_tanhf)
#    define TANH(x) __builtin_amdgcn_tanhf(x)   // v_tanh_f32 (CDNA5 trans op)
#  endif
#endif
#ifndef HAVE_ASYNC_LDS
#  define HAVE_ASYNC_LDS 0
#endif
#ifndef TANH
#  define TANH(x) tanhf(x)
#endif

#if HAVE_ASYNC_LDS
typedef __attribute__((address_space(1))) v4i as1_v4i;   // global
typedef __attribute__((address_space(3))) v4i as3_v4i;   // LDS
__device__ __forceinline__ void async_copy_b128(const void* g, void* l) {
  __builtin_amdgcn_global_load_async_to_lds_b128(
      (as1_v4i*)g, (as3_v4i*)l, 0, 0);
}
#endif

__device__ __forceinline__ unsigned short f32_to_bf16_rne(float f) {
  union { float f; unsigned u; } v; v.f = f;
  unsigned u = v.u;
  u += 0x7FFFu + ((u >> 16) & 1u);
  return (unsigned short)(u >> 16);
}

__device__ __forceinline__ v8f wmma_bf16(v16bf a, v16bf b, v8f c) {
  return __builtin_amdgcn_wmma_f32_16x16x32_bf16(
      false, a, false, b, (short)0, c, false, false);
}

#define P_PITCH 1032   // 1024 + 8 pad elements (kills LDS bank aliasing)

// Load the 4 B-tiles (32x16 bf16 each) for K-step kk: lane holds column l16,
// 16 contiguous K at kk*32 + half*16 -> one 32B read per tile from a bf16 W row.
__device__ __forceinline__ void load_b4(v16bf (&b)[4],
                                        const unsigned short* __restrict__ wrow,
                                        int kk) {
#pragma unroll
  for (int j = 0; j < 4; ++j)
    b[j] = *(const v16bf*)(wrow + (size_t)j * 16 * N_DIM + (kk << 5));
}

// Load the A-tile (16x32 bf16) for K-step kk from an LDS bf16 panel:
// lane holds row l16, 16 contiguous K at kk*32 + half*16.
__device__ __forceinline__ v16bf load_a(const __bf16 (*panel)[P_PITCH],
                                        int l16, int half, int kk) {
  const int ak = (kk << 5) + (half << 4);
  U32x8 ap;
  ap.lo = *(const u4x*)&panel[l16][ak];
  ap.hi = *(const u4x*)&panel[l16][ak + 8];
  return __builtin_bit_cast(v16bf, ap);
}

// ---------------------------------------------------------------------------
// Kernel 1: symmetrize W_self and convert both weight matrices to bf16.
// ---------------------------------------------------------------------------
__global__ void __launch_bounds__(256)
prep_weights(const float* __restrict__ Wl, const float* __restrict__ Wi,
             unsigned short* __restrict__ Wself_bf,
             unsigned short* __restrict__ Win_bf) {
  const int idx = blockIdx.x * 256 + threadIdx.x;
  const int i = idx >> 10, j = idx & (N_DIM - 1);
  const float ws = 0.5f * (Wl[idx] + Wl[j * N_DIM + i]);   // (W + W^T)/2
  Wself_bf[idx] = f32_to_bf16_rne(ws);
  Win_bf[idx]   = f32_to_bf16_rne(Wi[idx]);
}

// ---------------------------------------------------------------------------
// Kernel 2: x_proj = X(32768x1024) @ W_input^T -> Out (fp32)
// One block (16 waves) per 16-row M-tile:
//   - async-stage the fp32 X panel (16x1024) into LDS (ASYNCcnt path)
//   - convert once to a bf16 A-panel in LDS
//   - each wave owns 4 N-tiles; A from ds_load_b128, B double-buffered in regs
//     with a 2x-unrolled K loop (no inter-buffer register copies)
// ---------------------------------------------------------------------------
__global__ void __launch_bounds__(512)
xproj_gemm(const float* __restrict__ X,
           const unsigned short* __restrict__ Win_bf,
           float* __restrict__ Out) {
  __shared__ float  xs [16][P_PITCH];   // 66 KB fp32 staging
  __shared__ __bf16 abf[16][P_PITCH];   // 33 KB bf16 A panel

  const int tid  = threadIdx.x;
  const int lane = tid & 31;
  const int wave = tid >> 5;         // 0..15
  const int l16  = lane & 15;
  const int half = lane >> 4;
  const int mtile = blockIdx.x;      // 0..2047
  const int nbase = wave << 6;       // this wave's 64 output columns

  const float* xpanel = X + (size_t)mtile * 16 * N_DIM;

  // ---- stage X panel into LDS (b128 granules) ----
  for (int i = tid; i < 16 * (N_DIM / 4); i += 512) {
    const int row = i >> 8;              // 256 x b128 per row
    const int c4  = (i & 255) << 2;      // float column
#if HAVE_ASYNC_LDS
    async_copy_b128(xpanel + (size_t)row * N_DIM + c4, &xs[row][c4]);
#else
    *(float4*)&xs[row][c4] = *(const float4*)(xpanel + (size_t)row * N_DIM + c4);
#endif
  }
#if HAVE_ASYNC_LDS
  __builtin_amdgcn_s_wait_asynccnt(0);
#endif
  __syncthreads();

  // ---- one-time fp32 -> bf16 conversion of the A panel ----
  for (int i = tid; i < 16 * N_DIM; i += 512) {
    const int row = i >> 10, col = i & (N_DIM - 1);
    abf[row][col] = (__bf16)xs[row][col];
  }
  __syncthreads();

  // ---- GEMM: K=1024 in 32 steps, 2x-unrolled B ping-pong ----
  const unsigned short* wrow =
      Win_bf + (size_t)(nbase + l16) * N_DIM + (half << 4);

  v8f acc[4] = {v8f{}, v8f{}, v8f{}, v8f{}};
  v16bf ba[4], bb[4];
  load_b4(ba, wrow, 0);

  for (int kk = 0; kk < 32; kk += 2) {
    load_b4(bb, wrow, kk + 1);                       // prefetch odd step
    const v16bf a0 = load_a(abf, l16, half, kk);
#pragma unroll
    for (int j = 0; j < 4; ++j) acc[j] = wmma_bf16(a0, ba[j], acc[j]);

    if (kk + 2 < 32) load_b4(ba, wrow, kk + 2);      // prefetch next even step
    const v16bf a1 = load_a(abf, l16, half, kk + 1);
#pragma unroll
    for (int j = 0; j < 4; ++j) acc[j] = wmma_bf16(a1, bb[j], acc[j]);
  }

#pragma unroll
  for (int j = 0; j < 4; ++j) {
#pragma unroll
    for (int r = 0; r < 8; ++r) {
      const int row = (mtile << 4) + (half << 3) + r;
      Out[(size_t)row * N_DIM + nbase + (j << 4) + l16] = acc[j][r];
    }
  }
}

// ---------------------------------------------------------------------------
// Kernel 3: persistent scan. 4 workgroups x 512 threads (16 waves).
// Workgroup g owns batch rows [16g,16g+16) for all 512 timesteps; h state is
// a bf16 LDS panel re-fed into WMMA A-operands each step. Only workgroup
// barriers are needed (batch rows are independent across workgroups).
// ---------------------------------------------------------------------------
__global__ void __launch_bounds__(512)
rnn_scan(const unsigned short* __restrict__ Wself_bf,
         float* __restrict__ Out) {
  __shared__ __bf16 hbf[16][P_PITCH];   // 33 KB

  const int tid  = threadIdx.x;
  const int lane = tid & 31;
  const int wave = tid >> 5;        // 0..15
  const int l16  = lane & 15;
  const int half = lane >> 4;
  const int b0   = blockIdx.x << 4; // batch group base
  const int nb   = wave << 6;       // this wave's 64-column slab

  for (int i = tid; i < 16 * P_PITCH; i += 512)
    (&hbf[0][0])[i] = (__bf16)0.0f;                 // h(-1) = 0
  __syncthreads();

  const unsigned short* wrow =
      Wself_bf + (size_t)(nb + l16) * N_DIM + (half << 4);

  for (int t = 0; t < T_DIM; ++t) {
    v8f acc[4] = {v8f{}, v8f{}, v8f{}, v8f{}};
    v16bf ba[4], bb[4];
    load_b4(ba, wrow, 0);

    for (int kk = 0; kk < 32; kk += 2) {
      load_b4(bb, wrow, kk + 1);
      const v16bf a0 = load_a(hbf, l16, half, kk);
#pragma unroll
      for (int j = 0; j < 4; ++j) acc[j] = wmma_bf16(a0, ba[j], acc[j]);

      if (kk + 2 < 32) load_b4(ba, wrow, kk + 2);
      const v16bf a1 = load_a(hbf, l16, half, kk + 1);
#pragma unroll
      for (int j = 0; j < 4; ++j) acc[j] = wmma_bf16(a1, bb[j], acc[j]);
    }

    __syncthreads();   // all waves done reading h(t-1) from LDS

#pragma unroll
    for (int j = 0; j < 4; ++j) {
      const int ncol = nb + (j << 4) + l16;
#pragma unroll
      for (int r = 0; r < 8; ++r) {
        const int brow = (half << 3) + r;
        const size_t oidx =
            ((size_t)(b0 + brow) * T_DIM + t) * N_DIM + ncol;
        const float v = TANH(acc[j][r] + Out[oidx]);   // Out holds x_proj here
        Out[oidx] = v;                                  // overwrite with state
        hbf[brow][ncol] = (__bf16)v;
      }
    }

    __syncthreads();   // h(t) fully published before next step's reads
  }
}

// ---------------------------------------------------------------------------
extern "C" void kernel_launch(void* const* d_in, const int* in_sizes, int n_in,
                              void* d_out, int out_size, void* d_ws, size_t ws_size,
                              hipStream_t stream) {
  const float* X  = (const float*)d_in[0];   // (64, 512, 1024) fp32
  const float* Wl = (const float*)d_in[1];   // (1024, 1024)    fp32
  const float* Wi = (const float*)d_in[2];   // (1024, 1024)    fp32
  float* Out = (float*)d_out;                // (64, 512, 1024) fp32

  unsigned short* Wself_bf = (unsigned short*)d_ws;            // 2 MB
  unsigned short* Win_bf   = Wself_bf + (size_t)N_DIM * N_DIM; // 2 MB

  // 1) weights -> bf16 (W_self symmetrized)
  prep_weights<<<(N_DIM * N_DIM) / 256, 256, 0, stream>>>(Wl, Wi, Wself_bf, Win_bf);

  // 2) x_proj into Out: one block per 16-row M-tile (2048 blocks x 16 waves)
  xproj_gemm<<<2048, 512, 0, stream>>>(X, Win_bf, Out);

  // 3) persistent batch-parallel scan: 4 WGs x 16 waves
  rnn_scan<<<4, 512, 0, stream>>>(Wself_bf, Out);
}
